// SelfCrossAttention_86758339379620
// MI455X (gfx1250) — matrix operand order repaired and checked by reading, not compiled, not run on hardware
//
#include <hip/hip_runtime.h>
#include <hip/hip_bf16.h>
#include <cstdint>

// ---------------------------------------------------------------------------
// CDNA5 (gfx1250) WMMA types: 16x16x32 bf16 -> f32, wave32.
// ---------------------------------------------------------------------------
typedef __attribute__((ext_vector_type(16))) __bf16 v16bf;
typedef __attribute__((ext_vector_type(8)))  float  v8f;

union FragB16 { v16bf v; unsigned int u[8]; };

__device__ __forceinline__ v8f zero8() {
  v8f z = {0.f, 0.f, 0.f, 0.f, 0.f, 0.f, 0.f, 0.f};
  return z;
}

// round-to-nearest-even f32 -> bf16 (bit trick; avoids relying on __bf16 cvt)
__device__ __forceinline__ unsigned short f2bf(float f) {
  unsigned int u = __builtin_bit_cast(unsigned int, f);
  u += 0x7fffu + ((u >> 16) & 1u);
  return (unsigned short)(u >> 16);
}

// A-matrix 16x32 bf16 fragment K start for VGPR v, lane-half h:
// K = 2v + 16h (contiguous, matches B fragment loads)
__device__ __forceinline__ int a_kstart(int v, int half) {
  return (v << 1) + (half << 4);
}

__device__ __forceinline__ v8f wmma_bf16(const FragB16& a, const FragB16& b, v8f c) {
  return __builtin_amdgcn_wmma_f32_16x16x32_bf16(
      /*neg_a=*/false, a.v, /*neg_b=*/false, b.v,
      /*c_mod=*/(short)0, c, /*reuse_a=*/false, /*reuse_b=*/false);
}

// ---------------------------------------------------------------------------
// One wave computes a 32(M) x 64(N) tile of C = A(MxK,bf16,row-major) *
// B(KxN) where B is supplied TRANSPOSED: Bt[n][k] (pairs over k contiguous).
// Two A fragments (rows m0..m0+15 and m0+16..m0+31) share every B fragment,
// halving B-side L2 traffic per WMMA vs a 16-row tile.
// ---------------------------------------------------------------------------
__device__ __forceinline__ void gemm_tile_32x64(
    const unsigned short* __restrict__ A,   // points at row m0; leading dim lda
    int lda,
    const unsigned short* __restrict__ Bt,  // points at row n0 of Bt[N][K]
    int ldb,
    int K, v8f acc[2][4])
{
  const int lane = threadIdx.x & 31;
  const int nl   = lane & 15;
  const int half = lane >> 4;
  const unsigned short* arow0 = A + (size_t)nl * lda;
  const unsigned short* arow1 = A + (size_t)(16 + nl) * lda;
  for (int kc = 0; kc < K; kc += 32) {
    __builtin_prefetch(arow0 + kc + 64, 0, 1);  // global_prefetch_b8
    FragB16 a0, a1;
#pragma unroll
    for (int v = 0; v < 8; ++v) {
      a0.u[v] = *(const unsigned int*)(arow0 + kc + a_kstart(v, half));
      a1.u[v] = *(const unsigned int*)(arow1 + kc + a_kstart(v, half));
    }
#pragma unroll
    for (int f = 0; f < 4; ++f) {
      FragB16 b;
      const unsigned short* brow = Bt + (size_t)(f * 16 + nl) * ldb + kc + half * 16;
#pragma unroll
      for (int v = 0; v < 8; ++v)
        b.u[v] = *(const unsigned int*)(brow + v * 2);
      acc[0][f] = wmma_bf16(a0, b, acc[0][f]);
      acc[1][f] = wmma_bf16(a1, b, acc[1][f]);
    }
  }
}

// ---------------------------------------------------------------------------
// Elementwise converters
// ---------------------------------------------------------------------------
__global__ void cvt_f32_bf16_kernel(const float* __restrict__ src,
                                    unsigned short* __restrict__ dst, int n) {
  int i = blockIdx.x * blockDim.x + threadIdx.x;
  if (i < n) dst[i] = f2bf(src[i]);
}

// W is (K x N) row-major fp32; Wt is (N x K) bf16
__global__ void cvt_w_transpose_kernel(const float* __restrict__ W,
                                       unsigned short* __restrict__ Wt,
                                       int K, int N) {
  int i = blockIdx.x * blockDim.x + threadIdx.x;
  if (i < K * N) {
    int k = i / N, n = i % N;
    Wt[(size_t)n * K + k] = f2bf(W[i]);
  }
}

// tt[b][0:1024] = self_time[b], tt[b][1024:2048] = cross_time[b]
__global__ void pack_times_kernel(const int* __restrict__ st,
                                  const int* __restrict__ ct,
                                  int* __restrict__ tt) {
  int i = blockIdx.x * blockDim.x + threadIdx.x;  // B*2048
  int b = i >> 11, t = i & 2047;
  tt[i] = (t < 1024) ? st[b * 1024 + t] : ct[b * 1024 + (t - 1024)];
}

// ---------------------------------------------------------------------------
// QKV GEMM: X(4096x512,bf16) * W(512x1536) + bias, epilogue scatters into
// Qh/Kh (B*H, 2048, 64) bf16 and V transposed Vt (B*H, 64, 2048) bf16.
// Q is pre-scaled by 1/sqrt(64) = 0.125.
// ---------------------------------------------------------------------------
__global__ __launch_bounds__(256) void qkv_gemm_kernel(
    const unsigned short* __restrict__ X,
    const unsigned short* __restrict__ Wt,   // (1536 x 512) [n][k]
    const float* __restrict__ bias,          // 1536
    unsigned short* __restrict__ Qh,
    unsigned short* __restrict__ Kh,
    unsigned short* __restrict__ Vt,
    int toff)
{
  const int Kd = 512, Ntiles = 24;
  int wid = blockIdx.x * 8 + (threadIdx.x >> 5);
  int mtile = wid / Ntiles, ntile = wid % Ntiles;
  int m0 = mtile * 32, n0 = ntile * 64;

  v8f acc[2][4];
#pragma unroll
  for (int s = 0; s < 2; ++s)
#pragma unroll
    for (int f = 0; f < 4; ++f) acc[s][f] = zero8();
  gemm_tile_32x64(X + (size_t)m0 * Kd, Kd, Wt + (size_t)n0 * Kd, Kd, Kd, acc);

  const int lane = threadIdx.x & 31, nl = lane & 15, half = lane >> 4;
#pragma unroll
  for (int f = 0; f < 4; ++f) {
    int c  = n0 + f * 16 + nl;
    float bv = bias[c];
    int sec = c / 512, cc = c % 512, h = cc >> 6, ch = cc & 63;
#pragma unroll
    for (int s = 0; s < 2; ++s)
#pragma unroll
      for (int v = 0; v < 8; ++v) {
        int mrow = m0 + s * 16 + v + 8 * half;    // C frag: row = v + 8*half
        int b = mrow >> 10, tl = mrow & 1023;
        int t = toff + tl;
        float val = acc[s][f][v] + bv;
        size_t bh = (size_t)(b * 8 + h);
        if (sec == 0)      Qh[(bh * 2048 + t) * 64 + ch] = f2bf(val * 0.125f);
        else if (sec == 1) Kh[(bh * 2048 + t) * 64 + ch] = f2bf(val);
        else               Vt[(bh * 64 + ch) * 2048 + t] = f2bf(val);
      }
  }
}

// ---------------------------------------------------------------------------
// Flash attention: one wave owns 16 query rows, loops over 32-key tiles.
// 4 score WMMAs + 4 PV WMMAs per key tile; time-mask + online softmax.
// ---------------------------------------------------------------------------
__global__ __launch_bounds__(128) void attn_kernel(
    const unsigned short* __restrict__ Qh,
    const unsigned short* __restrict__ Kh,
    const unsigned short* __restrict__ Vt,
    const int* __restrict__ tt,
    unsigned short* __restrict__ Ybf)
{
  __shared__ unsigned short Plds[4][16][32];  // per-wave P transpose staging
  const int Ttot = 2048, hd = 64;
  const int lane = threadIdx.x & 31, w = threadIdx.x >> 5;
  const int nl = lane & 15, half = lane >> 4;
  const int qb = blockIdx.x & 31;     // 32 q-blocks of 64 rows
  const int bh = blockIdx.x >> 5;     // 0..31 (B*H)
  const int b = bh >> 3, h = bh & 7;
  const int q0 = qb * 64 + w * 16;

  const unsigned short* Qb = Qh + (size_t)bh * Ttot * hd;
  const unsigned short* Kb = Kh + (size_t)bh * Ttot * hd;
  const unsigned short* Vb = Vt + (size_t)bh * hd * Ttot;
  const int* tb = tt + b * Ttot;

  // Q A-fragments for hd chunks [0,32) and [32,64)
  FragB16 qa0, qa1;
  {
    const unsigned short* qrow = Qb + (size_t)(q0 + nl) * hd;
#pragma unroll
    for (int v = 0; v < 8; ++v) {
      qa0.u[v] = *(const unsigned int*)(qrow + a_kstart(v, half));
      qa1.u[v] = *(const unsigned int*)(qrow + 32 + a_kstart(v, half));
    }
  }
  int tq[8];
#pragma unroll
  for (int v = 0; v < 8; ++v) tq[v] = tb[q0 + v + 8 * half];

  float mrun[8], lrun[8];
  v8f acc[4];
#pragma unroll
  for (int v = 0; v < 8; ++v) { mrun[v] = -INFINITY; lrun[v] = 0.f; }
#pragma unroll
  for (int f = 0; f < 4; ++f) acc[f] = zero8();

  for (int kt = 0; kt < Ttot; kt += 32) {
    float S[2][8];
#pragma unroll
    for (int j = 0; j < 2; ++j) {
      FragB16 kb0, kb1;
      const unsigned short* krow =
          Kb + (size_t)(kt + j * 16 + nl) * hd + half * 16;
#pragma unroll
      for (int v = 0; v < 8; ++v) {
        kb0.u[v] = *(const unsigned int*)(krow + v * 2);
        kb1.u[v] = *(const unsigned int*)(krow + 32 + v * 2);
      }
      v8f c = zero8();
      c = wmma_bf16(qa0, kb0, c);
      c = wmma_bf16(qa1, kb1, c);
      int tk = tb[kt + j * 16 + nl];
#pragma unroll
      for (int v = 0; v < 8; ++v)
        S[j][v] = (tq[v] >= tk) ? c[v] : -1e30f;
    }

    // online softmax: row reductions across the 16-lane half-wave
    float mnew[8], scale[8], psum[8];
#pragma unroll
    for (int v = 0; v < 8; ++v) {
      float x = fmaxf(S[0][v], S[1][v]);
#pragma unroll
      for (int off = 1; off < 16; off <<= 1)
        x = fmaxf(x, __shfl_xor(x, off, 16));
      mnew[v]  = fmaxf(mrun[v], x);
      scale[v] = __expf(mrun[v] - mnew[v]);
      psum[v]  = 0.f;
    }
#pragma unroll
    for (int j = 0; j < 2; ++j)
#pragma unroll
      for (int v = 0; v < 8; ++v) {
        float p = __expf(S[j][v] - mnew[v]);
        psum[v] += p;
        Plds[w][v + 8 * half][j * 16 + nl] = f2bf(p);
      }
#pragma unroll
    for (int v = 0; v < 8; ++v) {
      float s = psum[v];
#pragma unroll
      for (int off = 1; off < 16; off <<= 1)
        s += __shfl_xor(s, off, 16);
      lrun[v] = lrun[v] * scale[v] + s;
      mrun[v] = mnew[v];
    }
#pragma unroll
    for (int f = 0; f < 4; ++f)
#pragma unroll
      for (int v = 0; v < 8; ++v)
        acc[f][v] *= scale[v];

    // same-wave LDS RAW: wait for the b16 stores before fragment reads
    asm volatile("s_wait_dscnt 0" ::: "memory");

    FragB16 pa;
    {
      const unsigned short* prow = &Plds[w][nl][0];
#pragma unroll
      for (int v = 0; v < 8; ++v)
        pa.u[v] = *(const unsigned int*)(prow + a_kstart(v, half));
    }
#pragma unroll
    for (int f = 0; f < 4; ++f) {
      FragB16 vb;
      const unsigned short* vcol =
          Vb + (size_t)(f * 16 + nl) * Ttot + kt + half * 16;
#pragma unroll
      for (int v = 0; v < 8; ++v)
        vb.u[v] = *(const unsigned int*)(vcol + v * 2);
      acc[f] = wmma_bf16(pa, vb, acc[f]);
    }
  }

#pragma unroll
  for (int f = 0; f < 4; ++f)
#pragma unroll
    for (int v = 0; v < 8; ++v) {
      float y = acc[f][v] / lrun[v];
      int row = q0 + v + 8 * half;
      Ybf[((size_t)b * Ttot + row) * 512 + h * 64 + f * 16 + nl] = f2bf(y);
    }
}

// ---------------------------------------------------------------------------
// Output projection: Y(8192x512,bf16) * W_proj + b, fp32 out with self/cross
// split ordering.
// ---------------------------------------------------------------------------
__global__ __launch_bounds__(256) void proj_gemm_kernel(
    const unsigned short* __restrict__ Ybf,   // 8192 x 512
    const unsigned short* __restrict__ Wpt,   // 512 x 512 [n][k]
    const float* __restrict__ bias,           // 512
    float* __restrict__ out)
{
  int wid = blockIdx.x * 8 + (threadIdx.x >> 5);
  int mtile = wid >> 3, ntile = wid & 7;
  int m0 = mtile * 32, n0 = ntile * 64;

  v8f acc[2][4];
#pragma unroll
  for (int s = 0; s < 2; ++s)
#pragma unroll
    for (int f = 0; f < 4; ++f) acc[s][f] = zero8();
  gemm_tile_32x64(Ybf + (size_t)m0 * 512, 512, Wpt + (size_t)n0 * 512, 512, 512, acc);

  const int lane = threadIdx.x & 31, nl = lane & 15, half = lane >> 4;
#pragma unroll
  for (int f = 0; f < 4; ++f) {
    int c = n0 + f * 16 + nl;
    float bv = bias[c];
#pragma unroll
    for (int s = 0; s < 2; ++s)
#pragma unroll
      for (int v = 0; v < 8; ++v) {
        int r = m0 + s * 16 + v + 8 * half;
        int b = r >> 11, t = r & 2047;
        float val = acc[s][f][v] + bv;
        size_t idx = (t < 1024)
            ? ((size_t)(b * 1024 + t) * 512 + c)
            : (size_t)(4 * 1024 * 512) + ((size_t)(b * 1024 + (t - 1024)) * 512 + c);
        out[idx] = val;
      }
  }
}

// ---------------------------------------------------------------------------
extern "C" void kernel_launch(void* const* d_in, const int* in_sizes, int n_in,
                              void* d_out, int out_size, void* d_ws, size_t ws_size,
                              hipStream_t stream) {
  (void)in_sizes; (void)n_in; (void)out_size; (void)ws_size;

  const float* self_seq  = (const float*)d_in[0];
  const float* cross_seq = (const float*)d_in[1];
  const int*   self_time = (const int*)d_in[2];
  const int*   cross_time= (const int*)d_in[3];
  const float* W_self    = (const float*)d_in[4];
  const float* b_self    = (const float*)d_in[5];
  const float* W_cross   = (const float*)d_in[6];
  const float* b_cross   = (const float*)d_in[7];
  const float* W_proj    = (const float*)d_in[8];
  const float* b_proj    = (const float*)d_in[9];
  float* out = (float*)d_out;

  char* ws = (char*)d_ws;
  size_t off = 0;
  auto alloc = [&](size_t bytes) {
    char* p = ws + off;
    off += (bytes + 255) & ~(size_t)255;
    return p;
  };
  unsigned short* Xs  = (unsigned short*)alloc(4096ull * 512 * 2);
  unsigned short* Xc  = (unsigned short*)alloc(4096ull * 512 * 2);
  unsigned short* Wst = (unsigned short*)alloc(1536ull * 512 * 2);
  unsigned short* Wct = (unsigned short*)alloc(1536ull * 512 * 2);
  unsigned short* Wpt = (unsigned short*)alloc(512ull * 512 * 2);
  unsigned short* Qh  = (unsigned short*)alloc(32ull * 2048 * 64 * 2);
  unsigned short* Kh  = (unsigned short*)alloc(32ull * 2048 * 64 * 2);
  unsigned short* Vt  = (unsigned short*)alloc(32ull * 64 * 2048 * 2);
  int*            tt  = (int*)alloc(4ull * 2048 * 4);
  unsigned short* Ybf = (unsigned short*)alloc(8192ull * 512 * 2);

  cvt_f32_bf16_kernel<<<8192, 256, 0, stream>>>(self_seq,  Xs, 4096 * 512);
  cvt_f32_bf16_kernel<<<8192, 256, 0, stream>>>(cross_seq, Xc, 4096 * 512);
  cvt_w_transpose_kernel<<<3072, 256, 0, stream>>>(W_self,  Wst, 512, 1536);
  cvt_w_transpose_kernel<<<3072, 256, 0, stream>>>(W_cross, Wct, 512, 1536);
  cvt_w_transpose_kernel<<<1024, 256, 0, stream>>>(W_proj,  Wpt, 512, 512);
  pack_times_kernel<<<32, 256, 0, stream>>>(self_time, cross_time, tt);

  // M=4096: 128 mtiles(32 rows) x 24 ntiles(64 cols) = 3072 waves -> 384 blocks
  qkv_gemm_kernel<<<384, 256, 0, stream>>>(Xs, Wst, b_self,  Qh, Kh, Vt, 0);
  qkv_gemm_kernel<<<384, 256, 0, stream>>>(Xc, Wct, b_cross, Qh, Kh, Vt, 1024);

  attn_kernel<<<1024, 128, 0, stream>>>(Qh, Kh, Vt, tt, Ybf);

  // M=8192: 256 mtiles(32 rows) x 8 ntiles(64 cols) = 2048 waves -> 256 blocks
  proj_gemm_kernel<<<256, 256, 0, stream>>>(Ybf, Wpt, b_proj, out);
}
